// LearnableVQ_15805479649603
// MI455X (gfx1250) — matrix operand order repaired and checked by reading, not compiled, not run on hardware
//
#include <hip/hip_runtime.h>

// ---------------------------------------------------------------------------
// Problem constants (from the reference)
// ---------------------------------------------------------------------------
#define EMB   768
#define NQ    1024
#define NDOC  1024
#define NNEG  4096
#define MSUB  32      // subvectors
#define KCODE 256     // codewords per subvector
#define DSUB  24      // dims per subvector
#define COLS  (NDOC + NNEG)   // 5120 score columns

// ---------------------------------------------------------------------------
// WMMA vector types (CDNA5 / gfx1250, wave32)
// ---------------------------------------------------------------------------
typedef __attribute__((ext_vector_type(16))) __bf16 v16bf;
typedef __attribute__((ext_vector_type(8)))  __bf16 v8bf;
typedef __attribute__((ext_vector_type(8)))  float  v8f;

union ABu { v16bf v; v8bf h[2]; };

// ---------------------------------------------------------------------------
// CDNA5 async global->LDS staging (ASYNCcnt path), with safe fallback.
// Builtin prototype (from hipcc diagnostic): first param is
//   addrspace(1) pointer to 'int __attribute__((vector_size(16)))'
// so we match it exactly. Flat->LDS address: the low 32 bits of the flat address
// (ISA 10.2), so integer truncation of a generic __shared__ pointer yields
// the LDS byte address for the addrspace(3) operand.
// ---------------------------------------------------------------------------
#if defined(__gfx1250__) && __has_builtin(__builtin_amdgcn_global_load_async_to_lds_b128)
#define VQ_ASYNC_LDS 1
#else
#define VQ_ASYNC_LDS 0
#endif

typedef int vq_v4i __attribute__((vector_size(16)));
typedef __attribute__((address_space(1))) vq_v4i* vq_gptr;   // global int4*
typedef __attribute__((address_space(3))) vq_v4i* vq_lptr;   // LDS int4*

__device__ __forceinline__ void vq_stage16(const __bf16* g, __bf16* l) {
#if VQ_ASYNC_LDS
    __builtin_amdgcn_global_load_async_to_lds_b128(
        (vq_gptr)(unsigned long long)(const void*)g,
        (vq_lptr)(unsigned)(unsigned long long)(void*)l,
        0, 0);
#else
    *(uint4*)l = *(const uint4*)g;
#endif
}

__device__ __forceinline__ void vq_async_drain() {
#if VQ_ASYNC_LDS
    asm volatile("s_wait_asynccnt 0x0" ::: "memory");
#endif
}

// ---------------------------------------------------------------------------
// Utility kernels
// ---------------------------------------------------------------------------
__global__ void vq_zero3(float* out) {
    if (threadIdx.x < 3) out[threadIdx.x] = 0.0f;
}

__global__ void vq_cvt_bf16(const float* __restrict__ x, __bf16* __restrict__ y, int n) {
    int i = blockIdx.x * blockDim.x + threadIdx.x;
    int stride = gridDim.x * blockDim.x;
    for (; i < n; i += stride) y[i] = (__bf16)x[i];
}

// Gather IVF centers; emit f32 (residuals / +center) and bf16 (scoring GEMM).
__global__ void vq_gather_centers(const float* __restrict__ centers,
                                  const int* __restrict__ ids,
                                  float* __restrict__ outf,
                                  __bf16* __restrict__ outh) {
    const int row = blockIdx.x;
    const int id  = ids[row];
    const float* src = centers + (size_t)id * EMB;
    for (int j = threadIdx.x; j < EMB; j += blockDim.x) {
        float v = src[j];
        outf[(size_t)row * EMB + j] = v;
        outh[(size_t)row * EMB + j] = (__bf16)v;
    }
}

// ---------------------------------------------------------------------------
// NT GEMM:  C[M,N](f32, ldc) = A[M,K](bf16) * B[N,K](bf16)^T
// Optionally also emits bf16 C (Cb != nullptr) to fuse the cvt for rq/rd/rn.
// Block = 256 threads (8 waves). Block tile 64x256, wave tile 32x64
// (2x4 v_wmma_f32_16x16x32_bf16 per K-step). Double-buffered LDS staged via
// async global->LDS copies (ASYNCcnt) so copies overlap the WMMAs.
// Requires: M%64==0, N%256==0, K%32==0 (true for all uses here).
// ---------------------------------------------------------------------------
__global__ __launch_bounds__(256)
void vq_gemm_nt_bf16(const __bf16* __restrict__ A, const __bf16* __restrict__ B,
                     float* __restrict__ C, __bf16* __restrict__ Cb,
                     int M, int N, int Kd, int ldc) {
    __shared__ __align__(16) __bf16 As[2][64 * 32];
    __shared__ __align__(16) __bf16 Bs[2][256 * 32];

    const int tid    = threadIdx.x;
    const int lane   = tid & 31;
    const int wid    = tid >> 5;
    const int wave_m = wid >> 2;          // 0..1 -> 32 rows
    const int wave_n = wid & 3;           // 0..3 -> 64 cols
    const int hsel   = lane >> 4;         // lane half
    const int l16    = lane & 15;
    const int row0   = blockIdx.y * 64;
    const int col0   = blockIdx.x * 256;

    // Per-thread staging coordinates (fixed across K-steps)
    const int a_r = tid >> 2, a_s = tid & 3;          // A: 256 b128 segs

    auto stage = [&](int buf, int k0) {
        vq_stage16(&A[(size_t)(row0 + a_r) * Kd + k0 + a_s * 8],
                   &As[buf][a_r * 32 + a_s * 8]);
        #pragma unroll
        for (int it = 0; it < 4; ++it) {              // B: 1024 b128 segs
            const int seg = tid + it * 256;
            const int br = seg >> 2, bs = seg & 3;
            vq_stage16(&B[(size_t)(col0 + br) * Kd + k0 + bs * 8],
                       &Bs[buf][br * 32 + bs * 8]);
        }
    };

    const v8f zero8 = {0.f, 0.f, 0.f, 0.f, 0.f, 0.f, 0.f, 0.f};
    v8f acc[2][4];
    #pragma unroll
    for (int mi = 0; mi < 2; ++mi)
        #pragma unroll
        for (int ni = 0; ni < 4; ++ni) acc[mi][ni] = zero8;

    stage(0, 0);
    vq_async_drain();
    __syncthreads();

    const int nk = Kd >> 5;
    for (int ki = 0; ki < nk; ++ki) {
        const int cur = ki & 1;
        if (ki + 1 < nk) stage(cur ^ 1, (ki + 1) << 5);   // overlap with WMMAs

        // Fragments per ISA 7.12.2 16-bit layouts.
        v16bf afrag[2], bfrag[4];
        #pragma unroll
        for (int mi = 0; mi < 2; ++mi) {
            const int ar = wave_m * 32 + mi * 16 + l16;   // M row
            const __bf16* p = &As[cur][ar * 32 + hsel * 8];
            ABu u;
            u.h[0] = *(const v8bf*)(p);        // K = kh..kh+7
            u.h[1] = *(const v8bf*)(p + 16);   // K = kh+16..kh+23
            afrag[mi] = u.v;
        }
        #pragma unroll
        for (int ni = 0; ni < 4; ++ni) {
            const int br = wave_n * 64 + ni * 16 + l16;   // N row (B stored [N,K])
            const __bf16* p = &Bs[cur][br * 32 + hsel * 8];
            ABu u;
            u.h[0] = *(const v8bf*)(p);
            u.h[1] = *(const v8bf*)(p + 16);
            bfrag[ni] = u.v;
        }

        #pragma unroll
        for (int mi = 0; mi < 2; ++mi)
            #pragma unroll
            for (int ni = 0; ni < 4; ++ni)
                acc[mi][ni] = __builtin_amdgcn_wmma_f32_16x16x32_bf16(
                    false, afrag[mi], false, bfrag[ni],
                    (short)0, acc[mi][ni], false, false);

        vq_async_drain();
        __syncthreads();
    }

    // C/D layout: VGPR r -> M = r + hsel*8, N = l16. Dims are exact multiples
    // of the tile, so stores are branch-free with incremental row pointers.
    #pragma unroll
    for (int mi = 0; mi < 2; ++mi) {
        #pragma unroll
        for (int ni = 0; ni < 4; ++ni) {
            const int rbase = row0 + wave_m * 32 + mi * 16 + hsel * 8;
            const int cidx  = col0 + wave_n * 64 + ni * 16 + l16;
            float* cp = C + (size_t)rbase * ldc + cidx;
            #pragma unroll
            for (int r = 0; r < 8; ++r) { *cp = acc[mi][ni][r]; cp += ldc; }
        }
    }
    if (Cb) {
        #pragma unroll
        for (int mi = 0; mi < 2; ++mi) {
            #pragma unroll
            for (int ni = 0; ni < 4; ++ni) {
                const int rbase = row0 + wave_m * 32 + mi * 16 + hsel * 8;
                const int cidx  = col0 + wave_n * 64 + ni * 16 + l16;
                __bf16* bp = Cb + (size_t)rbase * ldc + cidx;
                #pragma unroll
                for (int r = 0; r < 8; ++r) { *bp = (__bf16)acc[mi][ni][r]; bp += ldc; }
            }
        }
    }
}

// ---------------------------------------------------------------------------
// PQ assignment (forward of the STE quantizer = nearest codeword) fused with
// "+ center" and bf16 output for the scoring GEMM.
//   grid = (MSUB, rows/256), block = 256 (8 waves, 32 rows per wave)
//   dist_k = ||c_k||^2 - 2 * v.c_k   (||v||^2 constant per comparison)
// Codebook staged transposed in LDS (cb_t[j][c]) so lanes hit distinct banks.
// ---------------------------------------------------------------------------
__global__ __launch_bounds__(256)
void vq_assign(const float* __restrict__ R,     // rotated rows [rows,EMB]
               const float* __restrict__ Cen,   // gathered centers [rows,EMB]
               const float* __restrict__ CB,    // codebook [MSUB,KCODE,DSUB]
               __bf16* __restrict__ Q16,        // out: quantized rows bf16
               int rows) {
    __shared__ float cb_t[DSUB * KCODE];   // [j][c]
    __shared__ float cn[KCODE];

    const int tid  = threadIdx.x;
    const int lane = tid & 31;
    const int wid  = tid >> 5;
    const int m    = blockIdx.x;

    for (int i = tid; i < KCODE * DSUB; i += 256) {
        const int c = i / DSUB, j = i % DSUB;
        cb_t[j * KCODE + c] = CB[(size_t)m * KCODE * DSUB + i];
    }
    __syncthreads();
    {   // per-code squared norms
        float s = 0.f;
        #pragma unroll
        for (int j = 0; j < DSUB; ++j) {
            const float x = cb_t[j * KCODE + tid];
            s += x * x;
        }
        cn[tid] = s;
    }
    __syncthreads();

    const int rowBase = blockIdx.y * 256 + wid * 32;
    for (int i = 0; i < 32; ++i) {
        const int row = rowBase + i;
        if (row >= rows) break;
        const float* rrow = &R[(size_t)row * EMB + m * DSUB];
        const float* crow = &Cen[(size_t)row * EMB + m * DSUB];
        float v[DSUB];
        #pragma unroll
        for (int j = 0; j < DSUB; ++j) v[j] = rrow[j] - crow[j];

        float best = 3.0e38f;
        int   bidx = 0;
        #pragma unroll
        for (int cc = 0; cc < 8; ++cc) {
            const int c = lane + cc * 32;          // ascending -> keep-first ties
            float dot = 0.f;
            #pragma unroll
            for (int j = 0; j < DSUB; ++j) dot += v[j] * cb_t[j * KCODE + c];
            const float s = cn[c] - 2.f * dot;
            if (s < best) { best = s; bidx = c; }
        }
        // wave32 argmin with lowest-index tie-break (matches jnp.argmax-first)
        #pragma unroll
        for (int off = 16; off > 0; off >>= 1) {
            const float ob = __shfl_xor(best, off, 32);
            const int   oi = __shfl_xor(bidx, off, 32);
            if (ob < best || (ob == best && oi < bidx)) { best = ob; bidx = oi; }
        }
        if (lane < DSUB) {
            const float o = cb_t[lane * KCODE + bidx] + crow[lane];
            Q16[(size_t)row * EMB + m * DSUB + lane] = (__bf16)o;
        }
    }
}

// ---------------------------------------------------------------------------
// Distill: loss += mean_rows( -sum_j softmax(T)_j * log(softmax(S)_j + 1e-6) )
// One block per row, three passes over the (L2-resident) row.
// ---------------------------------------------------------------------------
__device__ __forceinline__ float vq_block_reduce(float v, float* red, bool domax) {
    const int tid = threadIdx.x;
    red[tid] = v;
    __syncthreads();
    for (int off = 128; off > 0; off >>= 1) {
        if (tid < off)
            red[tid] = domax ? fmaxf(red[tid], red[tid + off]) : (red[tid] + red[tid + off]);
        __syncthreads();
    }
    const float r = red[0];
    __syncthreads();
    return r;
}

__global__ __launch_bounds__(256)
void vq_distill(const float* __restrict__ T, const float* __restrict__ S,
                float* __restrict__ out, int idx) {
    __shared__ float red[256];
    const int row = blockIdx.x;
    const int tid = threadIdx.x;
    const float* t = T + (size_t)row * COLS;
    const float* s = S + (size_t)row * COLS;

    float mt = -3.0e38f, ms = -3.0e38f;
    for (int j = tid; j < COLS; j += 256) {
        mt = fmaxf(mt, t[j]);
        ms = fmaxf(ms, s[j]);
    }
    mt = vq_block_reduce(mt, red, true);
    ms = vq_block_reduce(ms, red, true);

    float zt = 0.f, zs = 0.f;
    for (int j = tid; j < COLS; j += 256) {
        zt += __expf(t[j] - mt);
        zs += __expf(s[j] - ms);
    }
    zt = vq_block_reduce(zt, red, false);
    zs = vq_block_reduce(zs, red, false);
    const float izt = 1.f / zt, izs = 1.f / zs;

    float ce = 0.f;
    for (int j = tid; j < COLS; j += 256) {
        const float tp = __expf(t[j] - mt) * izt;
        const float sp = __expf(s[j] - ms) * izs + 1e-6f;
        ce += tp * __logf(sp);
    }
    ce = vq_block_reduce(ce, red, false);
    if (tid == 0) atomicAdd(&out[idx], -ce * (1.0f / (float)NQ));
}

// ---------------------------------------------------------------------------
// Host-side orchestration
// ---------------------------------------------------------------------------
static inline void launch_gemm(const __bf16* A, const __bf16* B, float* C,
                               __bf16* Cb, int M, int N, int Kd, int ldc,
                               hipStream_t st) {
    dim3 grid(N / 256, M / 64), block(256);
    vq_gemm_nt_bf16<<<grid, block, 0, st>>>(A, B, C, Cb, M, N, Kd, ldc);
}

static inline void launch_cvt(const float* x, __bf16* y, int n, hipStream_t st) {
    int blocks = (n + 1023) / 1024;
    if (blocks > 2048) blocks = 2048;
    vq_cvt_bf16<<<blocks, 256, 0, st>>>(x, y, n);
}

extern "C" void kernel_launch(void* const* d_in, const int* in_sizes, int n_in,
                              void* d_out, int out_size, void* d_ws, size_t ws_size,
                              hipStream_t stream) {
    (void)in_sizes; (void)n_in; (void)out_size; (void)ws_size;

    const float* q_f   = (const float*)d_in[0];   // [1024,768]
    const float* d_f   = (const float*)d_in[1];   // [1024,768]
    const float* n_f   = (const float*)d_in[2];   // [4096,768]
    const float* rot_f = (const float*)d_in[3];   // [768,768]
    const float* cbk   = (const float*)d_in[4];   // [32,256,24]
    const float* ctr   = (const float*)d_in[5];   // [10000,768]
    const int*   did   = (const int*)d_in[6];     // [1024]
    const int*   nid   = (const int*)d_in[7];     // [4096]
    float*       out   = (float*)d_out;           // [3]

    // Workspace carve-out (256B aligned), ~113 MB total.
    char* wsb = (char*)d_ws;
    size_t off = 0;
    auto carve = [&](size_t bytes) -> char* {
        char* p = wsb + off;
        off = (off + bytes + 255) & ~(size_t)255;
        return p;
    };
    __bf16* rot16 = (__bf16*)carve((size_t)EMB * EMB * 2);
    __bf16* q16   = (__bf16*)carve((size_t)NQ   * EMB * 2);
    __bf16* d16   = (__bf16*)carve((size_t)NDOC * EMB * 2);
    __bf16* n16   = (__bf16*)carve((size_t)NNEG * EMB * 2);
    float*  rq    = (float*) carve((size_t)NQ   * EMB * 4);
    float*  rd    = (float*) carve((size_t)NDOC * EMB * 4);
    float*  rn    = (float*) carve((size_t)NNEG * EMB * 4);
    __bf16* rq16  = (__bf16*)carve((size_t)NQ   * EMB * 2);
    __bf16* rd16  = (__bf16*)carve((size_t)NDOC * EMB * 2);
    __bf16* rn16  = (__bf16*)carve((size_t)NNEG * EMB * 2);
    float*  dc    = (float*) carve((size_t)NDOC * EMB * 4);
    float*  nc    = (float*) carve((size_t)NNEG * EMB * 4);
    __bf16* dc16  = (__bf16*)carve((size_t)NDOC * EMB * 2);
    __bf16* nc16  = (__bf16*)carve((size_t)NNEG * EMB * 2);
    __bf16* qd16  = (__bf16*)carve((size_t)NDOC * EMB * 2);
    __bf16* qn16  = (__bf16*)carve((size_t)NNEG * EMB * 2);
    float*  Tsc   = (float*) carve((size_t)NQ * COLS * 4);  // teacher scores
    float*  Ssc   = (float*) carve((size_t)NQ * COLS * 4);  // student scores (reused)

    vq_zero3<<<1, 32, 0, stream>>>(out);

    // bf16 copies of raw inputs
    launch_cvt(rot_f, rot16, EMB * EMB, stream);
    launch_cvt(q_f,   q16,   NQ   * EMB, stream);
    launch_cvt(d_f,   d16,   NDOC * EMB, stream);
    launch_cvt(n_f,   n16,   NNEG * EMB, stream);

    // OPQ rotations: r? = ? @ rotate^T (NT GEMM); bf16 copy fused in epilogue
    launch_gemm(q16, rot16, rq, rq16, NQ,   EMB, EMB, EMB, stream);
    launch_gemm(d16, rot16, rd, rd16, NDOC, EMB, EMB, EMB, stream);
    launch_gemm(n16, rot16, rn, rn16, NNEG, EMB, EMB, EMB, stream);

    // IVF centroid gathers
    vq_gather_centers<<<NDOC, 256, 0, stream>>>(ctr, did, dc, dc16);
    vq_gather_centers<<<NNEG, 256, 0, stream>>>(ctr, nid, nc, nc16);

    // PQ on residuals (forward of STE = nearest codeword) + center add
    vq_assign<<<dim3(MSUB, NDOC / 256), 256, 0, stream>>>(rd, dc, cbk, qd16, NDOC);
    vq_assign<<<dim3(MSUB, NNEG / 256), 256, 0, stream>>>(rn, nc, cbk, qn16, NNEG);

    // Teacher scores: [q@d.T , q@n.T]
    launch_gemm(q16, d16, Tsc,        nullptr, NQ, NDOC, EMB, COLS, stream);
    launch_gemm(q16, n16, Tsc + NDOC, nullptr, NQ, NNEG, EMB, COLS, stream);

    // dense_score -> loss[0]
    launch_gemm(rq16, rd16, Ssc,        nullptr, NQ, NDOC, EMB, COLS, stream);
    launch_gemm(rq16, rn16, Ssc + NDOC, nullptr, NQ, NNEG, EMB, COLS, stream);
    vq_distill<<<NQ, 256, 0, stream>>>(Tsc, Ssc, out, 0);

    // ivf_score -> loss[1]
    launch_gemm(rq16, dc16, Ssc,        nullptr, NQ, NDOC, EMB, COLS, stream);
    launch_gemm(rq16, nc16, Ssc + NDOC, nullptr, NQ, NNEG, EMB, COLS, stream);
    vq_distill<<<NQ, 256, 0, stream>>>(Tsc, Ssc, out, 1);

    // pq_score -> loss[2]
    launch_gemm(rq16, qd16, Ssc,        nullptr, NQ, NDOC, EMB, COLS, stream);
    launch_gemm(rq16, qn16, Ssc + NDOC, nullptr, NQ, NNEG, EMB, COLS, stream);
    vq_distill<<<NQ, 256, 0, stream>>>(Tsc, Ssc, out, 2);
}
